// RNNTModel_21371757265241
// MI455X (gfx1250) — matrix operand order repaired and checked by reading, not compiled, not run on hardware
//
#include <hip/hip_runtime.h>

// RNN-T: B=8, T=256, U=64, D=80, H=P=256, V=1024, J=512
#define NB 8
#define NT 256
#define NU 64

typedef __attribute__((ext_vector_type(16))) __bf16 v16bf;
typedef __attribute__((ext_vector_type(8)))  float  v8f;
typedef __attribute__((ext_vector_type(4)))  unsigned int u32x4;
typedef __attribute__((ext_vector_type(8)))  int i32x8;
typedef __attribute__((ext_vector_type(4)))  int i32x4;

#if defined(__HIP_DEVICE_COMPILE__) && __has_builtin(__builtin_amdgcn_tensor_load_to_lds) && \
    __has_builtin(__builtin_amdgcn_s_wait_tensorcnt)
#define HAVE_TDM 1
#else
#define HAVE_TDM 0
#endif

__device__ __forceinline__ __bf16 f2bf(float f) {
  unsigned u = __builtin_bit_cast(unsigned, f);
  u += 0x7FFFu + ((u >> 16) & 1u);            // round-to-nearest-even
  unsigned short s = (unsigned short)(u >> 16);
  return __builtin_bit_cast(__bf16, s);
}

__device__ __forceinline__ float sigm(float x) { return 1.0f / (1.0f + __expf(-x)); }

// A-fragment (16x32 MxK, 16-bit): lane half selects K subset.
// vector idx 0..7 <-> K=k0..k0+7 ; idx 8..15 <-> K=k0+16..k0+23 ; k0 = ks*32 + half*8
__device__ __forceinline__ v16bf load_afrag(const __bf16* row, int k0) {
  union { u32x4 u[2]; v16bf v; } c;
  c.u[0] = *(const u32x4*)(row + k0);
  c.u[1] = *(const u32x4*)(row + k0 + 16);
  return c.v;
}

// B-fragment (32x16 KxN, 16-bit): lane%16 = N col; vector idx 0..7 <-> K=8*half+idx, 8..15 <-> K=16+8*half+idx-8 (Bt is K-major)
__device__ __forceinline__ v16bf load_bfrag(const __bf16* Bt, int ldb, int ks, int n0, int lane) {
  v16bf r;
  const __bf16* p = Bt + (size_t)(ks * 32 + (lane >> 4) * 8) * ldb + n0 + (lane & 15);
  #pragma unroll
  for (int s = 0; s < 16; ++s) r[s] = p[(size_t)((s < 8) ? s : s + 8) * ldb];
  return r;
}

__device__ __forceinline__ v8f wmma_bf16(v8f acc, v16bf a, v16bf b) {
  return __builtin_amdgcn_wmma_f32_16x16x32_bf16(false, a, false, b, (short)0, acc, false, false);
}

#if HAVE_TDM
// Issue a 2D TDM load: rows x 512 f32 tile, row stride 512 elems, into LDS.
__device__ __forceinline__ void tdm_load_rows512(const float* gsrc, const void* lds_dst,
                                                 unsigned rows) {
  unsigned long long ga = (unsigned long long)(uintptr_t)gsrc;
  unsigned lds_off = (unsigned)(size_t)(__attribute__((address_space(3))) const char*)lds_dst;
  u32x4 g0;
  g0[0] = 1u;                                         // count=1, user descriptor
  g0[1] = lds_off;                                    // lds_addr
  g0[2] = (unsigned)(ga & 0xFFFFFFFFu);               // global_addr[31:0]
  g0[3] = (unsigned)((ga >> 32) & 0x01FFFFFFu) | (2u << 30);   // global_addr[56:32] | type=2
  i32x8 g1;
  g1[0] = (int)(2u << 16);                            // data_size=4B, wg_mask=0
  g1[1] = (int)(512u << 16);                          // tensor_dim0[15:0]=512 (bits 63:48)
  g1[2] = (int)(rows << 16);                          // tensor_dim1[15:0]=rows (bits 95:80)
  g1[3] = (int)(512u << 16);                          // tile_dim0=512 (bits 127:112)
  g1[4] = (int)rows;                                  // tile_dim1=rows (bits 143:128)
  g1[5] = 512;                                        // tensor_dim0_stride[31:0]=512
  g1[6] = 0;
  g1[7] = 0;
  i32x4 gz4 = {0, 0, 0, 0};
  i32x8 gz8 = {0, 0, 0, 0, 0, 0, 0, 0};
  __builtin_amdgcn_tensor_load_to_lds(g0, g1, gz4, gz4, gz8, 0);
}
#endif

// ---------------------------------------------------------------- elementwise
__global__ void k_padcvt(const float* __restrict__ in, __bf16* __restrict__ out,
                         int M, int K, int Kpad) {
  int idx = blockIdx.x * blockDim.x + threadIdx.x;
  if (idx >= M * Kpad) return;
  int m = idx / Kpad, k = idx - m * Kpad;
  out[idx] = (k < K) ? f2bf(in[(size_t)m * K + k]) : f2bf(0.0f);
}

// f32 [N][ldin] (cols k=0..K-1) -> bf16 K-major [Kpad][N]
__global__ void k_tconv(const float* __restrict__ in, int ldin,
                        __bf16* __restrict__ out, int N, int K, int Kpad) {
  int idx = blockIdx.x * blockDim.x + threadIdx.x;
  if (idx >= Kpad * N) return;
  int k = idx / N, n = idx - k * N;
  out[idx] = (k < K) ? f2bf(in[(size_t)n * ldin + k]) : f2bf(0.0f);
}

__global__ void k_addvec(const float* __restrict__ a, const float* __restrict__ b,
                         float* __restrict__ o, int n) {
  int i = blockIdx.x * blockDim.x + threadIdx.x;
  if (i < n) o[i] = a[i] + b[i];
}

__global__ void k_gather(const float* __restrict__ emb, const int* __restrict__ tgt,
                         __bf16* __restrict__ out, int rows, int P) {
  int idx = blockIdx.x * blockDim.x + threadIdx.x;
  if (idx >= rows * P) return;
  int r = idx / P, p = idx - r * P;
  out[idx] = f2bf(emb[(size_t)tgt[r] * P + p]);
}

// ---------------------------------------------------------------- generic GEMM
// C[M][N] = A[M][K] * Bt(K-major [Kpad][ldb]) + bias.  M,N multiples of 64.
__global__ __launch_bounds__(128) void k_gemm(
    const __bf16* __restrict__ A, int lda,
    const __bf16* __restrict__ Bt, int ldb,
    const float* __restrict__ bias,
    float* __restrict__ C, int ldc, int Kpad) {
  int lane = threadIdx.x & 31, wave = threadIdx.x >> 5;
  int nn = lane & 15, hf = lane >> 4;
  int mbase = blockIdx.y * 64 + wave * 16;
  int nbase = blockIdx.x * 64;
  const __bf16* arow = A + (size_t)(mbase + nn) * lda;

  v8f acc[4] = {};
  int ksteps = Kpad >> 5;
  for (int ks = 0; ks < ksteps; ++ks) {
    v16bf a = load_afrag(arow, ks * 32 + hf * 8);
    #pragma unroll
    for (int nt = 0; nt < 4; ++nt) {
      v16bf b = load_bfrag(Bt, ldb, ks, nbase + nt * 16, lane);
      acc[nt] = wmma_bf16(acc[nt], a, b);
    }
  }
  #pragma unroll
  for (int nt = 0; nt < 4; ++nt) {
    int col = nbase + nt * 16 + nn;
    float bv = bias ? bias[col] : 0.0f;
    #pragma unroll
    for (int v = 0; v < 8; ++v) {
      int m = mbase + v + hf * 8;
      C[(size_t)m * ldc + col] = acc[nt][v] + bv;
    }
  }
}

// ---------------------------------------------------------------- LSTM scan
// One workgroup per direction. g = xg[:,t,:] + h @ whhT via WMMA (M=8 pad 16).
__global__ __launch_bounds__(256) void k_lstm(
    const float* __restrict__ xg, const __bf16* __restrict__ whhT,
    float* __restrict__ out, int Tsteps, int OS) {
  int dir = blockIdx.x;
  xg += (size_t)dir * 8 * Tsteps * 1024;
  whhT += (size_t)dir * 256 * 1024;
  int outOff = dir * 256;

  __shared__ __bf16 hbf[16][256];   // rows 8..15 stay zero (M padding)
  __shared__ float  cst[8][256];
  __shared__ float  gbuf[8][1024];

  for (int i = threadIdx.x; i < 16 * 256; i += 256) ((__bf16*)hbf)[i] = f2bf(0.0f);
  for (int i = threadIdx.x; i < 8 * 256; i += 256) ((float*)cst)[i] = 0.0f;
  __syncthreads();

  int lane = threadIdx.x & 31, wave = threadIdx.x >> 5;
  int nn = lane & 15, hf = lane >> 4;

  for (int step = 0; step < Tsteps; ++step) {
    int t = (dir == 1) ? (Tsteps - 1 - step) : step;
    for (int i = 0; i < 8; ++i) {
      int n0 = (wave + i * 8) * 16;
      v8f acc;
      #pragma unroll
      for (int v = 0; v < 8; ++v) {
        int m = v + hf * 8;
        acc[v] = (m < 8) ? xg[((size_t)m * Tsteps + t) * 1024 + n0 + nn] : 0.0f;
      }
      #pragma unroll
      for (int ks = 0; ks < 8; ++ks) {
        v16bf a = load_afrag(&hbf[nn][0], ks * 32 + hf * 8);
        v16bf b = load_bfrag(whhT, 1024, ks, n0, lane);
        acc = wmma_bf16(acc, a, b);
      }
      #pragma unroll
      for (int v = 0; v < 8; ++v) {
        int m = v + hf * 8;
        if (m < 8) gbuf[m][n0 + nn] = acc[v];
      }
    }
    __syncthreads();
    for (int r = 0; r < 8; ++r) {
      int idx = threadIdx.x + r * 256;
      int b = idx >> 8, j = idx & 255;
      float ig = gbuf[b][j], fg = gbuf[b][256 + j];
      float gg = gbuf[b][512 + j], og = gbuf[b][768 + j];
      float c = sigm(fg) * cst[b][j] + sigm(ig) * tanhf(gg);
      cst[b][j] = c;
      float h = sigm(og) * tanhf(c);
      hbf[b][j] = f2bf(h);
      out[((size_t)b * Tsteps + t) * OS + outOff + j] = h;
    }
    __syncthreads();
  }
}

// ---------------------------------------------------------------- fused joint
// One block per (b,t):  h[u][j] = relu(ep[bt][j] + pp[b*64+u][j]) -> bf16 LDS (all 64 u),
// logits[bt][u][v] = h @ w2T + b2.  pp tiles staged into LDS via TDM (double-buffered).
// Dynamic LDS: hb 64KB | epl 2KB | stage 2x32KB = 130KB
#define JNT_LDS_BYTES (64 * 512 * 2 + 512 * 4 + 2 * 16 * 512 * 4)
__global__ __launch_bounds__(256) void k_joint(
    const float* __restrict__ ep,   // [B*T][512]
    const float* __restrict__ pp,   // [B*U][512]
    const __bf16* __restrict__ w2T, // [512][1024] K-major
    const float* __restrict__ b2,   // [1024]
    float* __restrict__ out) {      // [B*T][U][V]
  extern __shared__ char smem[];
  __bf16* hb   = (__bf16*)smem;                       // [64][512]
  float*  epl  = (float*)(smem + 64 * 512 * 2);       // [512]
  float*  stg  = (float*)(smem + 64 * 512 * 2 + 512 * 4);  // [2][16][512]

  int bt = blockIdx.x;
  int b  = bt >> 8;                 // T = 256
  const float* ppb = pp + (size_t)b * 64 * 512;

  for (int i = threadIdx.x; i < 512; i += 256) epl[i] = ep[(size_t)bt * 512 + i];

  int lane = threadIdx.x & 31, wave = threadIdx.x >> 5;
  int nn = lane & 15, hf = lane >> 4;

#if HAVE_TDM
  // --- TDM-staged, double-buffered build of hb (4 chunks of 16 u-rows)
  if (wave == 0) tdm_load_rows512(ppb, stg, 16);
  for (int c = 0; c < 4; ++c) {
    if (wave == 0) {
      if (c < 3) {
        tdm_load_rows512(ppb + (size_t)(c + 1) * 16 * 512,
                         stg + ((c + 1) & 1) * 16 * 512, 16);
        __builtin_amdgcn_s_wait_tensorcnt(1);   // oldest DMA (chunk c) done
      } else {
        __builtin_amdgcn_s_wait_tensorcnt(0);   // last chunk done
      }
    }
    __syncthreads();                           // chunk c resident in stg[c&1]
    const float* sb = stg + (c & 1) * 16 * 512;
    for (int idx = threadIdx.x; idx < 16 * 512; idx += 256) {
      int j = idx & 511;
      float v = epl[j] + sb[idx];
      hb[(size_t)c * 16 * 512 + idx] = f2bf(v > 0.0f ? v : 0.0f);
    }
    __syncthreads();                           // stg[c&1] free for chunk c+2
  }
#else
  __syncthreads();                             // epl ready
  for (int idx = threadIdx.x; idx < 64 * 512; idx += 256) {
    int j = idx & 511;
    float v = epl[j] + ppb[idx];
    hb[idx] = f2bf(v > 0.0f ? v : 0.0f);
  }
  __syncthreads();
#endif

  // --- GEMM: 4 m-tiles (u) x 64 n-tiles (v); each wave 8 n-tiles.
  float* obt = out + (size_t)bt * 64 * 1024;
  for (int i = 0; i < 8; ++i) {
    int n0 = (wave + i * 8) * 16;
    float b2v = b2[n0 + nn];
    v8f acc[4] = {};
    for (int ks = 0; ks < 16; ++ks) {
      __builtin_prefetch((const void*)(w2T + (size_t)(ks * 32 + 32 + lane) * 1024 + n0), 0, 1);
      v16bf bf = load_bfrag(w2T, 1024, ks, n0, lane);
      #pragma unroll
      for (int mt = 0; mt < 4; ++mt) {
        v16bf a = load_afrag(hb + (size_t)(mt * 16 + nn) * 512, ks * 32 + hf * 8);
        acc[mt] = wmma_bf16(acc[mt], a, bf);
      }
    }
    #pragma unroll
    for (int mt = 0; mt < 4; ++mt)
      #pragma unroll
      for (int v = 0; v < 8; ++v) {
        int u = mt * 16 + v + hf * 8;
        obt[(size_t)u * 1024 + n0 + nn] = acc[mt][v] + b2v;
      }
  }
}

// ---------------------------------------------------------------- host
static inline dim3 g1(long n) { return dim3((unsigned)((n + 255) / 256)); }

extern "C" void kernel_launch(void* const* d_in, const int* in_sizes, int n_in,
                              void* d_out, int out_size, void* d_ws, size_t ws_size,
                              hipStream_t stream) {
  const float* feats    = (const float*)d_in[0];   // [8,256,80]
  const int*   targets  = (const int*)d_in[1];     // [8,64]
  const float* e0_wih   = (const float*)d_in[2];   // [2,1024,80]
  const float* e0_whh   = (const float*)d_in[3];   // [2,1024,256]
  const float* e0_bih   = (const float*)d_in[4];
  const float* e0_bhh   = (const float*)d_in[5];
  const float* e1_wih   = (const float*)d_in[6];   // [2,1024,512]
  const float* e1_whh   = (const float*)d_in[7];
  const float* e1_bih   = (const float*)d_in[8];
  const float* e1_bhh   = (const float*)d_in[9];
  const float* emb      = (const float*)d_in[10];  // [1024,256]
  const float* p_wih    = (const float*)d_in[11];  // [1024,256]
  const float* p_whh    = (const float*)d_in[12];
  const float* p_bih    = (const float*)d_in[13];
  const float* p_bhh    = (const float*)d_in[14];
  const float* w1       = (const float*)d_in[15];  // [512,768]
  const float* b1       = (const float*)d_in[16];
  const float* w2       = (const float*)d_in[17];  // [1024,512]
  const float* b2       = (const float*)d_in[18];
  float* out = (float*)d_out;

  char* W = (char*)d_ws;
  size_t off = 0;
  auto carve = [&](size_t bytes) -> char* {
    char* p = W + off; off += (bytes + 255) & ~(size_t)255; return p;
  };
  const int MBT = NB * NT;   // 2048
  const int MBU = NB * NU;   // 512

  __bf16* xin_bf  = (__bf16*)carve((size_t)MBT * 96 * 2);
  __bf16* wihT0   = (__bf16*)carve((size_t)2 * 96 * 1024 * 2);
  __bf16* whhT0   = (__bf16*)carve((size_t)2 * 256 * 1024 * 2);
  __bf16* wihT1   = (__bf16*)carve((size_t)2 * 512 * 1024 * 2);
  __bf16* whhT1   = (__bf16*)carve((size_t)2 * 256 * 1024 * 2);
  __bf16* pwihT   = (__bf16*)carve((size_t)256 * 1024 * 2);
  __bf16* pwhhT   = (__bf16*)carve((size_t)256 * 1024 * 2);
  __bf16* w1encT  = (__bf16*)carve((size_t)512 * 512 * 2);
  __bf16* w1predT = (__bf16*)carve((size_t)256 * 512 * 2);
  __bf16* w2T     = (__bf16*)carve((size_t)512 * 1024 * 2);
  float*  bias0   = (float*)carve((size_t)2 * 1024 * 4);
  float*  bias1   = (float*)carve((size_t)2 * 1024 * 4);
  float*  biasp   = (float*)carve((size_t)1024 * 4);
  float*  xg      = (float*)carve((size_t)2 * MBT * 1024 * 4);  // reused layer0/1
  float*  e0_out  = (float*)carve((size_t)MBT * 512 * 4);
  __bf16* e0_bf   = (__bf16*)carve((size_t)MBT * 512 * 2);
  float*  e1_out  = (float*)carve((size_t)MBT * 512 * 4);
  __bf16* e1_bf   = (__bf16*)carve((size_t)MBT * 512 * 2);
  __bf16* px_bf   = (__bf16*)carve((size_t)MBU * 256 * 2);
  float*  xgp     = (float*)carve((size_t)MBU * 1024 * 4);
  float*  p_out   = (float*)carve((size_t)MBU * 256 * 4);
  __bf16* p_bf    = (__bf16*)carve((size_t)MBU * 256 * 2);
  float*  epj     = (float*)carve((size_t)MBT * 512 * 4);
  float*  ppj     = (float*)carve((size_t)MBU * 512 * 4);
  (void)ws_size; (void)in_sizes; (void)n_in; (void)out_size;

  // ---- weight conversion (bf16, K-major for B operands)
  k_padcvt<<<g1((long)MBT * 96), 256, 0, stream>>>(feats, xin_bf, MBT, 80, 96);
  for (int d = 0; d < 2; ++d) {
    k_tconv<<<g1(96L * 1024), 256, 0, stream>>>(e0_wih + (size_t)d * 1024 * 80, 80,
                                                wihT0 + (size_t)d * 96 * 1024, 1024, 80, 96);
    k_tconv<<<g1(256L * 1024), 256, 0, stream>>>(e0_whh + (size_t)d * 1024 * 256, 256,
                                                 whhT0 + (size_t)d * 256 * 1024, 1024, 256, 256);
    k_tconv<<<g1(512L * 1024), 256, 0, stream>>>(e1_wih + (size_t)d * 1024 * 512, 512,
                                                 wihT1 + (size_t)d * 512 * 1024, 1024, 512, 512);
    k_tconv<<<g1(256L * 1024), 256, 0, stream>>>(e1_whh + (size_t)d * 1024 * 256, 256,
                                                 whhT1 + (size_t)d * 256 * 1024, 1024, 256, 256);
    k_addvec<<<g1(1024), 256, 0, stream>>>(e0_bih + d * 1024, e0_bhh + d * 1024, bias0 + d * 1024, 1024);
    k_addvec<<<g1(1024), 256, 0, stream>>>(e1_bih + d * 1024, e1_bhh + d * 1024, bias1 + d * 1024, 1024);
  }
  k_tconv<<<g1(256L * 1024), 256, 0, stream>>>(p_wih, 256, pwihT, 1024, 256, 256);
  k_tconv<<<g1(256L * 1024), 256, 0, stream>>>(p_whh, 256, pwhhT, 1024, 256, 256);
  k_addvec<<<g1(1024), 256, 0, stream>>>(p_bih, p_bhh, biasp, 1024);
  k_tconv<<<g1(512L * 512), 256, 0, stream>>>(w1, 768, w1encT, 512, 512, 512);
  k_tconv<<<g1(256L * 512), 256, 0, stream>>>(w1 + 512, 768, w1predT, 512, 256, 256);
  k_tconv<<<g1(512L * 1024), 256, 0, stream>>>(w2, 512, w2T, 1024, 512, 512);

  // ---- encoder layer 0
  for (int d = 0; d < 2; ++d)
    k_gemm<<<dim3(1024 / 64, MBT / 64), 128, 0, stream>>>(
        xin_bf, 96, wihT0 + (size_t)d * 96 * 1024, 1024,
        bias0 + d * 1024, xg + (size_t)d * MBT * 1024, 1024, 96);
  k_lstm<<<2, 256, 0, stream>>>(xg, whhT0, e0_out, NT, 512);
  k_padcvt<<<g1((long)MBT * 512), 256, 0, stream>>>(e0_out, e0_bf, MBT, 512, 512);

  // ---- encoder layer 1
  for (int d = 0; d < 2; ++d)
    k_gemm<<<dim3(1024 / 64, MBT / 64), 128, 0, stream>>>(
        e0_bf, 512, wihT1 + (size_t)d * 512 * 1024, 1024,
        bias1 + d * 1024, xg + (size_t)d * MBT * 1024, 1024, 512);
  k_lstm<<<2, 256, 0, stream>>>(xg, whhT1, e1_out, NT, 512);
  k_padcvt<<<g1((long)MBT * 512), 256, 0, stream>>>(e1_out, e1_bf, MBT, 512, 512);

  // ---- prediction network
  k_gather<<<g1((long)MBU * 256), 256, 0, stream>>>(emb, targets, px_bf, MBU, 256);
  k_gemm<<<dim3(1024 / 64, MBU / 64), 128, 0, stream>>>(
      px_bf, 256, pwihT, 1024, biasp, xgp, 1024, 256);
  k_lstm<<<1, 256, 0, stream>>>(xgp, pwhhT, p_out, NU, 256);
  k_padcvt<<<g1((long)MBU * 256), 256, 0, stream>>>(p_out, p_bf, MBU, 256, 256);

  // ---- joint projections (b1 folded into enc side)
  k_gemm<<<dim3(512 / 64, MBT / 64), 128, 0, stream>>>(
      e1_bf, 512, w1encT, 512, b1, epj, 512, 512);
  k_gemm<<<dim3(512 / 64, MBU / 64), 128, 0, stream>>>(
      p_bf, 256, w1predT, 512, nullptr, ppj, 512, 256);

  // ---- fused joint + logits GEMM (dominant: 137 GFLOP)
  k_joint<<<MBT, 256, JNT_LDS_BYTES, stream>>>(epj, ppj, w2T, b2, out);
}
